// MoEWithDeepGEMM_62569083568927
// MI455X (gfx1250) — compile-verified, matrix operand order checked by reading, not run on hardware
//
#include <hip/hip_runtime.h>
#include <hip/hip_bf16.h>
#include <stdint.h>

// Problem sizes (fixed by the reference).
#define M_ 4096
#define D_ 2048
#define H_ 2048
#define E_ 8

typedef unsigned long long ull;
typedef __attribute__((ext_vector_type(16))) int   v16i;
typedef __attribute__((ext_vector_type(4)))  int   av4i;
typedef __attribute__((ext_vector_type(8)))  float v8f;

// ---- feature probes (guarded so the file compiles on any toolchain) ----
#if defined(__HIP_DEVICE_COMPILE__) && __has_builtin(__builtin_amdgcn_global_load_async_to_lds_b128)
#define HAVE_ASYNC_LDS 1
#else
#define HAVE_ASYNC_LDS 0
#endif

#if defined(__HIP_DEVICE_COMPILE__) && __has_builtin(__builtin_amdgcn_cvt_pk_fp8_f32)
#define HAVE_CVT_FP8 1
#else
#define HAVE_CVT_FP8 0
#endif

// 16-byte global->LDS copy; async TDM-style path when available.
// gfx1250 builtin signature: (v4i AS1* src, v4i AS3* dst, imm offset, imm cpol)
__device__ __forceinline__ void async_copy16(unsigned char* lds_dst,
                                             const unsigned char* gsrc) {
#if HAVE_ASYNC_LDS
  __builtin_amdgcn_global_load_async_to_lds_b128(
      (__attribute__((address_space(1))) av4i*)(void*)gsrc,
      (__attribute__((address_space(3))) av4i*)(void*)lds_dst,
      0, 0);
#else
  *(int4*)lds_dst = *(const int4*)gsrc;
#endif
}

__device__ __forceinline__ void wait_async() {
#if HAVE_ASYNC_LDS
#if __has_builtin(__builtin_amdgcn_s_wait_asynccnt)
  __builtin_amdgcn_s_wait_asynccnt(0);
#else
  asm volatile("s_wait_asynccnt 0" ::: "memory");
#endif
#endif
}

// f32 -> OCP e4m3fn, round-to-nearest-even, saturating, denormal-correct.
__device__ __forceinline__ unsigned f32_to_e4m3_sw(float x) {
  unsigned b = __float_as_uint(x);
  unsigned s = (b >> 24) & 0x80u;
  unsigned a = b & 0x7FFFFFFFu;
  if (a >= 0x7F800000u) return s | 0x7Fu;   // Inf/NaN -> NaN
  if (a >= 0x43E00000u) return s | 0x7Eu;   // |x| >= 448 -> max (448)
  if (a < 0x3C800000u) {                    // |x| < 2^-6: denormal fixed-point
    float f = __uint_as_float(a) * 512.0f;  // units of 2^-9 (e4m3 denorm ulp)
    unsigned q = (unsigned)__float2int_rn(f);
    return s | q;                           // 0..8 (8 == min normal, continuous)
  }
  unsigned exp = a >> 23;
  unsigned mant = a & 0x7FFFFFu;
  unsigned lsb = (mant >> 20) & 1u;
  mant += 0x7FFFFu + lsb;                   // RTNE at bit 20
  if (mant >> 23) { mant = 0; exp += 1; }
  unsigned m3 = (mant >> 20) & 7u;
  unsigned code = ((exp - 120u) << 3) | m3; // e8 = exp - 127 + 7
  if (code > 0x7Eu) code = 0x7Eu;
  return s | code;
}

// f32 -> bf16 (RTNE) -> f32, emulating the reference's .astype(bf16).
__device__ __forceinline__ float bf16_rne(float x) {
  unsigned u = __float_as_uint(x);
  unsigned lsb = (u >> 16) & 1u;
  u += 0x7FFFu + lsb;
  u &= 0xFFFF0000u;
  return __uint_as_float(u);
}

// ---------------- Kernel 1: gating (logits + softmax over E=8) -------------
__global__ __launch_bounds__(256) void gate_softmax_kernel(
    const float* __restrict__ x, const float* __restrict__ gw,
    const float* __restrict__ gb, float* __restrict__ gates) {
  const int lane = threadIdx.x & 31;
  const int m = blockIdx.x * 8 + (threadIdx.x >> 5);   // one wave32 per token
  float acc[E_];
#pragma unroll
  for (int e = 0; e < E_; ++e) acc[e] = 0.f;
  const float* xr = x + (size_t)m * D_;
  for (int d = lane; d < D_; d += 32) {
    float xv = xr[d];
#pragma unroll
    for (int e = 0; e < E_; ++e) acc[e] += xv * gw[(size_t)e * D_ + d];
  }
#pragma unroll
  for (int off = 16; off > 0; off >>= 1) {
#pragma unroll
    for (int e = 0; e < E_; ++e) acc[e] += __shfl_xor(acc[e], off, 32);
  }
  float l[E_], mx = -3.4e38f;
#pragma unroll
  for (int e = 0; e < E_; ++e) { l[e] = acc[e] + gb[e]; mx = fmaxf(mx, l[e]); }
  float ssum = 0.f;
#pragma unroll
  for (int e = 0; e < E_; ++e) { l[e] = __expf(l[e] - mx); ssum += l[e]; }
  float inv = 1.f / ssum;
  if (lane < E_) gates[(size_t)m * E_ + lane] = l[lane] * inv;
}

// ---------------- Kernel 2: f32 -> fp8 e4m3 quantize (pack 4/thread) -------
__global__ __launch_bounds__(256) void quant4_kernel(
    const float* __restrict__ src, unsigned int* __restrict__ dst, int n4) {
  int i = blockIdx.x * blockDim.x + threadIdx.x;
  if (i >= n4) return;
  float4 v = ((const float4*)src)[i];
  unsigned p;
#if HAVE_CVT_FP8
  int lo = __builtin_amdgcn_cvt_pk_fp8_f32(v.x, v.y, 0, false);
  p = (unsigned)__builtin_amdgcn_cvt_pk_fp8_f32(v.z, v.w, lo, true);
#else
  p = f32_to_e4m3_sw(v.x) | (f32_to_e4m3_sw(v.y) << 8) |
      (f32_to_e4m3_sw(v.z) << 16) | (f32_to_e4m3_sw(v.w) << 24);
#endif
  dst[i] = p;
}

// Load a 16x128 fp8 A/B fragment from an LDS tile stored row-major with
// 128-byte rows.  ISA 8-bit layout: lane (m + 16h) holds 8-byte K-chunks at
// byte offsets 8h + 16q (q=0..3) and +64, into VGPR dword pairs (2q,2q+1)
// and (8+2q, 8+2q+1).  `base` must already include row*128 + 8*half.
__device__ __forceinline__ v16i load_frag128(const unsigned char* base) {
  union { v16i v; ull u[8]; } f;
#pragma unroll
  for (int q = 0; q < 4; ++q) {
    f.u[q]     = *(const ull*)(base + 16 * q);
    f.u[4 + q] = *(const ull*)(base + 64 + 16 * q);
  }
  return f.v;
}

// ---------------- Kernel 3: fused 8-expert fp8 GEMM + bf16 cast + combine --
// Block: 256 threads (8 waves, 4M x 2N), output tile 128(M) x 128(H).
// Wave tile 32x64 -> 2x4 = 8 v_wmma_f32_16x16x128_fp8_fp8 per k-step
// (24 ds_load_2addr_b64 per 8 WMMAs -> matrix-pipe bound).
// Per expert: accumulate over D=2048 in 16 k-steps, then
// out += gate[m,e] * bf16(acc).
__global__ __launch_bounds__(256) void moe_fp8_gemm_kernel(
    const unsigned char* __restrict__ xq,   // [M, D] fp8
    const unsigned char* __restrict__ wq,   // [E, H, D] fp8
    const float* __restrict__ gates,        // [M, E]
    float* __restrict__ out) {              // [M, H] f32
  __shared__ unsigned char Abuf[2][128 * 128];  // 2 x 16 KB
  __shared__ unsigned char Bbuf[2][128 * 128];  // 2 x 16 KB  (total 64 KB)

  const int tid  = threadIdx.x;
  const int lane = tid & 31;
  const int wid  = tid >> 5;
  const int wm   = wid & 3;    // wave row 0..3   (32 rows each)
  const int wn   = wid >> 2;   // wave col 0..1   (64 cols each)
  const int half = lane >> 4;  // lane half (ISA "h")
  const int l16  = lane & 15;
  const int mb   = blockIdx.x * 128;
  const int hb   = blockIdx.y * 128;

  const v8f vzero = {0.f, 0.f, 0.f, 0.f, 0.f, 0.f, 0.f, 0.f};
  v8f outacc[2][4];
#pragma unroll
  for (int mf = 0; mf < 2; ++mf)
#pragma unroll
    for (int nf = 0; nf < 4; ++nf) outacc[mf][nf] = vzero;

  auto stage = [&](int kb, size_t wrow_base, int buf) {
    // A tile: 128 rows x 128 B = 1024 16-byte chunks; 4 per thread.
#pragma unroll
    for (int i = 0; i < 4; ++i) {
      int c = tid + 256 * i;
      int row = c >> 3, col = (c & 7) << 4;
      async_copy16(&Abuf[buf][row * 128 + col],
                   xq + (size_t)(mb + row) * D_ + kb + col);
    }
    // B tile: 128 rows x 128 B = 1024 chunks; 4 per thread.
#pragma unroll
    for (int i = 0; i < 4; ++i) {
      int c = tid + 256 * i;
      int row = c >> 3, col = (c & 7) << 4;
      async_copy16(&Bbuf[buf][row * 128 + col],
                   wq + wrow_base + (size_t)row * D_ + kb + col);
    }
  };

  for (int e = 0; e < E_; ++e) {
    const size_t wrow_base = ((size_t)e * H_ + hb) * D_;
    v8f acc[2][4];
#pragma unroll
    for (int mf = 0; mf < 2; ++mf)
#pragma unroll
      for (int nf = 0; nf < 4; ++nf) acc[mf][nf] = vzero;

    stage(0, wrow_base, 0);
    wait_async();
    __syncthreads();

    for (int k = 0; k < D_ / 128; ++k) {
      const int cur = k & 1;
      if (k + 1 < D_ / 128) stage((k + 1) * 128, wrow_base, cur ^ 1);

      v16i a0 = load_frag128(&Abuf[cur][(wm * 32 + l16) * 128 + 8 * half]);
      v16i a1 = load_frag128(&Abuf[cur][(wm * 32 + 16 + l16) * 128 + 8 * half]);
#pragma unroll
      for (int nf = 0; nf < 4; ++nf) {
        v16i b = load_frag128(&Bbuf[cur][(wn * 64 + nf * 16 + l16) * 128 + 8 * half]);
        acc[0][nf] = __builtin_amdgcn_wmma_f32_16x16x128_fp8_fp8(
            a0, b, (short)0, acc[0][nf], false, false);
        acc[1][nf] = __builtin_amdgcn_wmma_f32_16x16x128_fp8_fp8(
            a1, b, (short)0, acc[1][nf], false, false);
      }
      wait_async();
      __syncthreads();
    }

    // Emulate bf16 cast of y[e], then gate-weighted combine in f32.
    // C/D layout: vgpr r, lane (n + 16h) -> element (m = r + 8h, n).
#pragma unroll
    for (int mf = 0; mf < 2; ++mf) {
      const int mrow = mb + wm * 32 + mf * 16 + half * 8;
#pragma unroll
      for (int r = 0; r < 8; ++r) {
        float g = gates[(size_t)(mrow + r) * E_ + e];   // L2-resident broadcast
#pragma unroll
        for (int nf = 0; nf < 4; ++nf)
          outacc[mf][nf][r] += g * bf16_rne(acc[mf][nf][r]);
      }
    }
  }

  // Store f32 output tile.
#pragma unroll
  for (int mf = 0; mf < 2; ++mf) {
    const int mrow = mb + wm * 32 + mf * 16 + half * 8;
#pragma unroll
    for (int nf = 0; nf < 4; ++nf) {
      const int n = hb + wn * 64 + nf * 16 + l16;
#pragma unroll
      for (int r = 0; r < 8; ++r)
        out[(size_t)(mrow + r) * H_ + n] = outacc[mf][nf][r];
    }
  }
}

// ---------------------------------------------------------------------------
extern "C" void kernel_launch(void* const* d_in, const int* in_sizes, int n_in,
                              void* d_out, int out_size, void* d_ws,
                              size_t ws_size, hipStream_t stream) {
  (void)in_sizes; (void)n_in; (void)out_size; (void)ws_size;
  const float* x        = (const float*)d_in[0];  // [M, D]
  const float* gate_w   = (const float*)d_in[1];  // [E, D]
  const float* gate_b   = (const float*)d_in[2];  // [E]
  const float* expert_w = (const float*)d_in[3];  // [E, H, D]
  float* out = (float*)d_out;                     // [M, H] f32

  // Workspace layout (~40.2 MB): gates | x_fp8 | w_fp8
  float* gates = (float*)d_ws;                                   // 128 KB
  unsigned char* xq = (unsigned char*)d_ws + (1u << 17);         // 8 MB
  unsigned char* wq = xq + (size_t)M_ * D_;                      // 32 MB

  // 1. Gating softmax: one wave per token.
  gate_softmax_kernel<<<M_ / 8, 256, 0, stream>>>(x, gate_w, gate_b, gates);

  // 2. fp8 e4m3 quantize of activations and expert weights.
  {
    int n4x = M_ * D_ / 4;
    quant4_kernel<<<(n4x + 255) / 256, 256, 0, stream>>>(x, (unsigned*)xq, n4x);
    int n4w = E_ * H_ * D_ / 4;
    quant4_kernel<<<(n4w + 255) / 256, 256, 0, stream>>>(expert_w, (unsigned*)wq, n4w);
  }

  // 3. Fused 8-expert fp8 WMMA GEMM + bf16 cast + gate combine.
  dim3 grid(M_ / 128, H_ / 128);
  moe_fp8_gemm_kernel<<<grid, 256, 0, stream>>>(xq, wq, gates, out);
}
